// MoELayer_76673756168498
// MI455X (gfx1250) — matrix operand order repaired and checked by reading, not compiled, not run on hardware
//
#include <hip/hip_runtime.h>

#define NTOK 8192
#define HID  1024
#define FFN  4096
#define NEXP 8
#define CAP  2560            // int(1.25 * 8192/8 * 2)
#define MT   128
#define NTILE 128
#define KT   64
#define STR  72              // LDS row stride (elems): 144B, 16B-aligned, conflict-free
#define ROWT (CAP/MT)        // 20
#define NT1  (FFN/NTILE)     // 32
#define NT2  (HID/NTILE)     // 8

typedef __attribute__((ext_vector_type(16))) __bf16 v16bf;
typedef __attribute__((ext_vector_type(8)))  float  v8f;
typedef __attribute__((ext_vector_type(4)))  unsigned int v4u;

union bfvec { v4u u[2]; v16bf v; };

// branchless tanh-form gelu: x * sigmoid(1.5957691*x*(1+0.044715*x^2))
__device__ __forceinline__ float gelu_fast(float x) {
    float u = 1.5957691216057308f * x * (1.0f + 0.044715f * x * x);
    float s = __builtin_amdgcn_rcpf(1.0f + __expf(-u));
    return x * s;
}

// ---- CDNA5 async memory->LDS copy (ASYNCcnt path, ISA 08_async_tensor §4) ----
__device__ __forceinline__ void cp_b128o(unsigned ldsOff, const __bf16* g) {
    asm volatile("global_load_async_to_lds_b128 %0, %1, off"
                 :: "v"(ldsOff), "v"((unsigned long long)(uintptr_t)g) : "memory");
}
__device__ __forceinline__ void wait_async() {
    asm volatile("s_wait_asynccnt 0x0" ::: "memory");
}

// ---------------- utility kernels ----------------

__global__ __launch_bounds__(256) void zero_f32(float* p, size_t n) {
    size_t i = (size_t)blockIdx.x * blockDim.x + threadIdx.x;
    size_t stride = (size_t)gridDim.x * blockDim.x;
    for (; i < n; i += stride) p[i] = 0.0f;
}

__global__ __launch_bounds__(256) void cvt_bf16(const float* __restrict__ in,
                                                __bf16* __restrict__ out, size_t n) {
    size_t i = (size_t)blockIdx.x * blockDim.x + threadIdx.x;
    size_t stride = (size_t)gridDim.x * blockDim.x;
    for (; i < n; i += stride) out[i] = (__bf16)in[i];
}

// transpose one [R][C] f32 slab per expert -> [C][R] bf16 (LDS-tiled, coalesced both sides)
__global__ __launch_bounds__(256)
void transpose_cvt(const float* __restrict__ in, __bf16* __restrict__ out, int R, int C) {
    int tilesC = C >> 5, tilesR = R >> 5;
    int bt = blockIdx.x;
    int tc = bt % tilesC;
    int tr = (bt / tilesC) % tilesR;
    int e  = bt / (tilesC * tilesR);
    const float* src = in + (size_t)e * R * C;
    __bf16* dst = out + (size_t)e * R * C;
    __shared__ __bf16 tile[32][33];
    int tid = threadIdx.x;
    int c = tid & 31, r0 = tid >> 5;
#pragma unroll
    for (int rr = 0; rr < 32; rr += 8)
        tile[r0 + rr][c] = (__bf16)src[(size_t)(tr*32 + r0 + rr) * C + tc*32 + c];
    __syncthreads();
#pragma unroll
    for (int rr = 0; rr < 32; rr += 8)
        dst[(size_t)(tc*32 + r0 + rr) * R + tr*32 + c] = tile[c][r0 + rr];
}

// ---------------- router: logits -> top2 ids + renormalized gates ----------------
__global__ __launch_bounds__(256)
void router_topk(const float* __restrict__ x, const float* __restrict__ rw,
                 int* __restrict__ eid, float* __restrict__ gate) {
    int wave = threadIdx.x >> 5, lane = threadIdx.x & 31;
    int t = blockIdx.x * 8 + wave;
    const float* xr = x + (size_t)t * HID;
    float p[NEXP];
#pragma unroll
    for (int e = 0; e < NEXP; ++e) p[e] = 0.0f;
    for (int h = lane; h < HID; h += 32) {
        float xv = xr[h];
#pragma unroll
        for (int e = 0; e < NEXP; ++e) p[e] += xv * rw[e * HID + h];
    }
#pragma unroll
    for (int e = 0; e < NEXP; ++e)
        for (int off = 16; off > 0; off >>= 1) p[e] += __shfl_xor(p[e], off, 32);
    if (lane == 0) {
        int i0 = 0; float v0 = p[0];
#pragma unroll
        for (int e = 1; e < NEXP; ++e) if (p[e] > v0) { v0 = p[e]; i0 = e; }
        int i1 = (i0 == 0) ? 1 : 0; float v1 = p[i1];
#pragma unroll
        for (int e = 0; e < NEXP; ++e)
            if (e != i0 && p[e] > v1) { v1 = p[e]; i1 = e; }
        float r = expf(v1 - v0);
        float g0 = 1.0f / (1.0f + r);
        eid[t] = i0; eid[NTOK + t] = i1;
        gate[t] = g0; gate[NTOK + t] = 1.0f - g0;
    }
}

// ---------------- ordered capacity scan (single workgroup, token order) ----------------
__global__ __launch_bounds__(256)
void scan_routes(const int* __restrict__ eid, const float* __restrict__ gate,
                 int* __restrict__ route, float* __restrict__ gslot,
                 int* __restrict__ counts) {
    __shared__ int run[2][NEXP];
    __shared__ int wcnt[2][NEXP][8];
    int tid = threadIdx.x, wave = tid >> 5, lane = tid & 31;
    if (tid < 16) run[tid >> 3][tid & 7] = 0;
    __syncthreads();
    unsigned ltmask = lane ? (0xFFFFFFFFu >> (32 - lane)) : 0u;
    for (int c = 0; c < NTOK / 256; ++c) {
        int t = c * 256 + tid;
        int myeid[2] = { eid[t], eid[NTOK + t] };
        int within[2] = { 0, 0 };
#pragma unroll
        for (int k = 0; k < 2; ++k) {
#pragma unroll
            for (int e = 0; e < NEXP; ++e) {
                unsigned m = (unsigned)__ballot(myeid[k] == e);
                if (myeid[k] == e) within[k] = __popc(m & ltmask);
                if (lane == 0) wcnt[k][e][wave] = __popc(m);
            }
        }
        __syncthreads();
#pragma unroll
        for (int k = 0; k < 2; ++k) {
            int e = myeid[k];
            int pos = run[k][e] + within[k];
            for (int w = 0; w < wave; ++w) pos += wcnt[k][e][w];
            if (pos < CAP) {
                int slot = (k * NEXP + e) * CAP + pos;
                route[slot] = t;
                gslot[slot] = gate[k * NTOK + t];
            }
        }
        __syncthreads();
        if (tid < 16) {
            int k = tid >> 3, e = tid & 7, s = 0;
#pragma unroll
            for (int w = 0; w < 8; ++w) s += wcnt[k][e][w];
            run[k][e] += s;
        }
        __syncthreads();
    }
    if (tid < 16) counts[tid] = run[tid >> 3][tid & 7];
}

// ---------------- GEMM1: H = gelu(X_gathered @ w1[e]) (bf16 out) ----------------
__global__ __launch_bounds__(256)
void moe_gemm1(const __bf16* __restrict__ xbf, const __bf16* __restrict__ w1t,
               const int* __restrict__ route, const int* __restrict__ counts,
               __bf16* __restrict__ hbuf, int kslot) {
    int nt = blockIdx.x % NT1;
    int rt = (blockIdx.x / NT1) % ROWT;
    int e  = blockIdx.x / (NT1 * ROWT);
    int cnt = counts[kslot * NEXP + e]; if (cnt > CAP) cnt = CAP;
    int mBase = rt * MT;
    if (mBase >= cnt) return;
    int nBase = nt * NTILE;
    const __bf16* Bw = w1t + (size_t)e * HID * FFN;   // [FFN][HID]
    const int* routeKE = route + (size_t)(kslot * NEXP + e) * CAP + mBase;

    __shared__ int rows[MT];
    __shared__ __align__(16) __bf16 Asm[2 * MT * STR];
    __shared__ __align__(16) __bf16 Bsm[2 * NTILE * STR];
    const unsigned ABUF = MT * STR * 2;      // bytes per A buffer
    const unsigned BBUF = NTILE * STR * 2;

    int tid = threadIdx.x;
    if (tid < MT) rows[tid] = (mBase + tid < cnt) ? routeKE[tid] : routeKE[0];
    __syncthreads();

    const __bf16* aG[4]; const __bf16* bG[4];
    unsigned aLo[4], bLo[4];
#pragma unroll
    for (int it = 0; it < 4; ++it) {
        int x = tid + 256 * it;
        int row = x >> 3, ch = (x & 7) * 8;
        aG[it] = xbf + (size_t)rows[row] * HID + ch;
        aLo[it] = (unsigned)(uintptr_t)&Asm[row * STR + ch];
        bG[it] = Bw + (size_t)(nBase + row) * HID + ch;
        bLo[it] = (unsigned)(uintptr_t)&Bsm[row * STR + ch];
    }

    int lane = tid & 31, wave = tid >> 5;
    int wm = wave & 3, wn = wave >> 2;
    int lh = lane & 15;
    int gA = (lane < 16) ? 0 : 8;
    int gB = (lane < 16) ? 0 : 8;
    int rowA[2], rowB[4];
#pragma unroll
    for (int i = 0; i < 2; ++i) rowA[i] = (wm*32 + i*16 + lh) * STR;
#pragma unroll
    for (int j = 0; j < 4; ++j) rowB[j] = (wn*64 + j*16 + lh) * STR;

    v8f acc[2][4] = {};

    auto issue = [&](int buf, int kb) {
#pragma unroll
        for (int it = 0; it < 4; ++it) {
            cp_b128o(aLo[it] + buf * ABUF, aG[it] + kb);
            cp_b128o(bLo[it] + buf * BBUF, bG[it] + kb);
        }
    };
    auto compute = [&](int buf) {
        const __bf16* Ab = &Asm[buf * MT * STR];
        const __bf16* Bb = &Bsm[buf * NTILE * STR];
#pragma unroll
        for (int f = 0; f < 2; ++f) {
            bfvec a[2], b[4];
#pragma unroll
            for (int i = 0; i < 2; ++i) {
                const __bf16* p = Ab + rowA[i] + f*32 + gA;
                a[i].u[0] = *(const v4u*)p;
                a[i].u[1] = *(const v4u*)(p + 16);
            }
#pragma unroll
            for (int j = 0; j < 4; ++j) {
                const __bf16* p = Bb + rowB[j] + f*32 + gB;
                b[j].u[0] = *(const v4u*)p;
                b[j].u[1] = *(const v4u*)(p + 16);
            }
#pragma unroll
            for (int i = 0; i < 2; ++i)
#pragma unroll
                for (int j = 0; j < 4; ++j)
                    acc[i][j] = __builtin_amdgcn_wmma_f32_16x16x32_bf16(
                        false, a[i].v, false, b[j].v, (short)0, acc[i][j], false, false);
        }
    };

    // double-buffered async pipeline: copies for tile t+1 overlap compute of tile t
    issue(0, 0);
    const int NITER = HID / KT;
#pragma unroll 2
    for (int t2 = 0; t2 < NITER; ++t2) {
        int buf = t2 & 1;
        wait_async();
        __syncthreads();
        if (t2 + 1 < NITER) issue(buf ^ 1, (t2 + 1) * KT);
        compute(buf);
    }

    size_t hrowBase = (size_t)e * CAP + mBase;
#pragma unroll
    for (int i = 0; i < 2; ++i)
#pragma unroll
        for (int j = 0; j < 4; ++j) {
            int n = nBase + wn * 64 + j * 16 + lh;
#pragma unroll
            for (int v = 0; v < 8; ++v) {
                int ml = wm * 32 + i * 16 + v + ((lane < 16) ? 0 : 8);
                hbuf[(hrowBase + ml) * FFN + n] = (__bf16)gelu_fast(acc[i][j][v]);
            }
        }
}

// ---------------- GEMM2: out[token] += gate * (H @ w2[e]) ----------------
__global__ __launch_bounds__(256)
void moe_gemm2(const __bf16* __restrict__ hbuf, const __bf16* __restrict__ w2t,
               const int* __restrict__ route, const float* __restrict__ gslot,
               const int* __restrict__ counts, float* __restrict__ out, int kslot) {
    int nt = blockIdx.x % NT2;
    int rt = (blockIdx.x / NT2) % ROWT;
    int e  = blockIdx.x / (NT2 * ROWT);
    int cnt = counts[kslot * NEXP + e]; if (cnt > CAP) cnt = CAP;
    int mBase = rt * MT;
    if (mBase >= cnt) return;
    int nBase = nt * NTILE;
    const __bf16* Bw = w2t + (size_t)e * FFN * HID;   // [HID][FFN]
    size_t slabBase = (size_t)(kslot * NEXP + e) * CAP + mBase;

    __shared__ int rows[MT];
    __shared__ float gts[MT];
    __shared__ __align__(16) __bf16 Asm[2 * MT * STR];
    __shared__ __align__(16) __bf16 Bsm[2 * NTILE * STR];
    const unsigned ABUF = MT * STR * 2;
    const unsigned BBUF = NTILE * STR * 2;

    int tid = threadIdx.x;
    if (tid < MT) {
        bool vld = (mBase + tid) < cnt;
        rows[tid] = vld ? route[slabBase + tid] : 0;
        gts[tid]  = vld ? gslot[slabBase + tid] : 0.0f;
    }
    __syncthreads();

    const __bf16* aG[4]; const __bf16* bG[4];
    unsigned aLo[4], bLo[4];
#pragma unroll
    for (int it = 0; it < 4; ++it) {
        int x = tid + 256 * it;
        int row = x >> 3, ch = (x & 7) * 8;
        aG[it] = hbuf + ((size_t)e * CAP + mBase + row) * FFN + ch;
        aLo[it] = (unsigned)(uintptr_t)&Asm[row * STR + ch];
        bG[it] = Bw + (size_t)(nBase + row) * FFN + ch;
        bLo[it] = (unsigned)(uintptr_t)&Bsm[row * STR + ch];
    }

    int lane = tid & 31, wave = tid >> 5;
    int wm = wave & 3, wn = wave >> 2;
    int lh = lane & 15;
    int gA = (lane < 16) ? 0 : 8;
    int gB = (lane < 16) ? 0 : 8;
    int rowA[2], rowB[4];
#pragma unroll
    for (int i = 0; i < 2; ++i) rowA[i] = (wm*32 + i*16 + lh) * STR;
#pragma unroll
    for (int j = 0; j < 4; ++j) rowB[j] = (wn*64 + j*16 + lh) * STR;

    v8f acc[2][4] = {};

    auto issue = [&](int buf, int kb) {
#pragma unroll
        for (int it = 0; it < 4; ++it) {
            cp_b128o(aLo[it] + buf * ABUF, aG[it] + kb);
            cp_b128o(bLo[it] + buf * BBUF, bG[it] + kb);
        }
    };
    auto compute = [&](int buf) {
        const __bf16* Ab = &Asm[buf * MT * STR];
        const __bf16* Bb = &Bsm[buf * NTILE * STR];
#pragma unroll
        for (int f = 0; f < 2; ++f) {
            bfvec a[2], b[4];
#pragma unroll
            for (int i = 0; i < 2; ++i) {
                const __bf16* p = Ab + rowA[i] + f*32 + gA;
                a[i].u[0] = *(const v4u*)p;
                a[i].u[1] = *(const v4u*)(p + 16);
            }
#pragma unroll
            for (int j = 0; j < 4; ++j) {
                const __bf16* p = Bb + rowB[j] + f*32 + gB;
                b[j].u[0] = *(const v4u*)p;
                b[j].u[1] = *(const v4u*)(p + 16);
            }
#pragma unroll
            for (int i = 0; i < 2; ++i)
#pragma unroll
                for (int j = 0; j < 4; ++j)
                    acc[i][j] = __builtin_amdgcn_wmma_f32_16x16x32_bf16(
                        false, a[i].v, false, b[j].v, (short)0, acc[i][j], false, false);
        }
    };

    issue(0, 0);
    const int NITER = FFN / KT;
#pragma unroll 2
    for (int t2 = 0; t2 < NITER; ++t2) {
        int buf = t2 & 1;
        wait_async();
        __syncthreads();
        if (t2 + 1 < NITER) issue(buf ^ 1, (t2 + 1) * KT);
        compute(buf);
    }

#pragma unroll
    for (int i = 0; i < 2; ++i)
#pragma unroll
        for (int j = 0; j < 4; ++j) {
            int n = nBase + wn * 64 + j * 16 + lh;
#pragma unroll
            for (int v = 0; v < 8; ++v) {
                int ml = wm * 32 + i * 16 + v + ((lane < 16) ? 0 : 8);
                if (mBase + ml < cnt) {
                    int tok = rows[ml];
                    atomicAdd(&out[(size_t)tok * HID + n], gts[ml] * acc[i][j][v]);
                }
            }
        }
}

// ---------------- launch ----------------
extern "C" void kernel_launch(void* const* d_in, const int* in_sizes, int n_in,
                              void* d_out, int out_size, void* d_ws, size_t ws_size,
                              hipStream_t stream) {
    const float* x  = (const float*)d_in[0];
    const float* rw = (const float*)d_in[1];
    const float* w1 = (const float*)d_in[2];
    const float* w2 = (const float*)d_in[3];
    float* out = (float*)d_out;

    char* ws = (char*)d_ws;
    size_t off = 0;
    auto alloc = [&](size_t bytes) {
        size_t o = off; off += (bytes + 255) & ~(size_t)255; return o;
    };
    __bf16* xbf   = (__bf16*)(ws + alloc((size_t)NTOK * HID * 2));
    __bf16* w1t   = (__bf16*)(ws + alloc((size_t)NEXP * HID * FFN * 2));
    __bf16* w2t   = (__bf16*)(ws + alloc((size_t)NEXP * FFN * HID * 2));
    __bf16* hbuf  = (__bf16*)(ws + alloc((size_t)NEXP * CAP * FFN * 2));  // reused per k
    int*    eid   = (int*)  (ws + alloc((size_t)2 * NTOK * 4));
    float*  gate  = (float*)(ws + alloc((size_t)2 * NTOK * 4));
    int*    route = (int*)  (ws + alloc((size_t)2 * NEXP * CAP * 4));
    float*  gslot = (float*)(ws + alloc((size_t)2 * NEXP * CAP * 4));
    int*    counts= (int*)  (ws + alloc(64));
    (void)ws_size; (void)in_sizes; (void)n_in; (void)out_size;

    zero_f32<<<2048, 256, 0, stream>>>(out, (size_t)NTOK * HID);
    cvt_bf16<<<2048, 256, 0, stream>>>(x, xbf, (size_t)NTOK * HID);
    transpose_cvt<<<(FFN/32)*(HID/32)*NEXP, 256, 0, stream>>>(w1, w1t, HID, FFN);
    transpose_cvt<<<(HID/32)*(FFN/32)*NEXP, 256, 0, stream>>>(w2, w2t, FFN, HID);
    router_topk<<<NTOK / 8, 256, 0, stream>>>(x, rw, eid, gate);
    scan_routes<<<1, 256, 0, stream>>>(eid, gate, route, gslot, counts);
    for (int k = 0; k < 2; ++k) {
        moe_gemm1<<<NEXP * ROWT * NT1, 256, 0, stream>>>(xbf, w1t, route, counts, hbuf, k);
        moe_gemm2<<<NEXP * ROWT * NT2, 256, 0, stream>>>(hbuf, w2t, route, gslot, counts, out, k);
    }
}
